// MoELayer_38543036514484
// MI455X (gfx1250) — matrix operand order repaired and checked by reading, not compiled, not run on hardware
//
#include <hip/hip_runtime.h>
#include <hip/hip_bf16.h>

#define HIDDEN 1024
#define NUM_EXPERTS 16
#define TOPK 6
#define NTOK 8192           // 4 * 2048
#define EPS 1e-10f

typedef _Float16 v16h __attribute__((ext_vector_type(16)));
typedef _Float16 v8h  __attribute__((ext_vector_type(8)));
typedef float    v8f  __attribute__((ext_vector_type(8)));

// ---------------------------------------------------------------- utilities
__global__ void zero_counts_kernel(int* counts) {
    if (threadIdx.x < NUM_EXPERTS) counts[threadIdx.x] = 0;
}

// ------------------------------------------------------------- router logits
__global__ void router_logits_kernel(const float* __restrict__ tokens,
                                     const float* __restrict__ rw,
                                     const float* __restrict__ rb,
                                     float* __restrict__ logits) {
    int g = blockIdx.x * blockDim.x + threadIdx.x;   // g = t*16 + e
    int t = g >> 4;
    int e = g & 15;
    const float4* x = (const float4*)(tokens + (size_t)t * HIDDEN);
    const float4* w = (const float4*)(rw + (size_t)e * HIDDEN);
    float s = 0.f;
#pragma unroll 4
    for (int i = 0; i < HIDDEN / 4; ++i) {
        float4 a = x[i];
        float4 b = w[i];
        s += a.x * b.x + a.y * b.y + a.z * b.z + a.w * b.w;
    }
    logits[g] = s + rb[e];
}

// -------------------------------------------- softmax + top-k + scatter route
__global__ void route_kernel(const float* __restrict__ logits,
                             float* __restrict__ addend,
                             int* __restrict__ counts,
                             int* __restrict__ tok_list,
                             float* __restrict__ wt_list) {
    int t = blockIdx.x * blockDim.x + threadIdx.x;
    if (t >= NTOK) return;

    float l[NUM_EXPERTS];
    float m = -3.4e38f;
#pragma unroll
    for (int e = 0; e < NUM_EXPERTS; ++e) {
        l[e] = logits[t * NUM_EXPERTS + e];
        m = fmaxf(m, l[e]);
    }
    float p[NUM_EXPERTS];
    float s = 0.f;
#pragma unroll
    for (int e = 0; e < NUM_EXPERTS; ++e) {
        p[e] = __expf(l[e] - m);
        s += p[e];
    }
    float inv = 1.f / s;
#pragma unroll
    for (int e = 0; e < NUM_EXPERTS; ++e) p[e] *= inv;

    int   idx[TOPK];
    float tp[TOPK];
    float S = 0.f;
#pragma unroll
    for (int k = 0; k < TOPK; ++k) {
        int best = 0;
        float bv = -1.f;
#pragma unroll
        for (int e = 0; e < NUM_EXPERTS; ++e) {
            if (p[e] > bv) { bv = p[e]; best = e; }
        }
        idx[k] = best;
        tp[k]  = bv;
        p[best] = -1.f;
        S += bv;
    }
    float rn    = 1.f / (S + EPS);
    float acc   = S * rn;                  // sum of renormalized top-k
    float scale = 1.f / fmaxf(acc, EPS);   // folds the /acc into the pair weight
    addend[t]   = EPS * acc;               // the "+ EPS * sum(top_p)" term

#pragma unroll
    for (int k = 0; k < TOPK; ++k) {
        int e   = idx[k];
        int pos = atomicAdd(&counts[e], 1);
        tok_list[e * NTOK + pos] = t;
        wt_list[e * NTOK + pos]  = tp[k] * rn * scale;
    }
}

// ----------------------------------------------------- init output to addend
__global__ void init_out_kernel(float* __restrict__ out,
                                const float* __restrict__ addend) {
    int g = blockIdx.x * blockDim.x + threadIdx.x;
    out[g] = addend[g >> 10];   // H == 1024
}

// --------------------------------------------------- tokens f32 -> f16 copy
__global__ void conv_tokens_kernel(const float* __restrict__ x,
                                   _Float16* __restrict__ xh) {
    size_t g = (size_t)(blockIdx.x * blockDim.x + threadIdx.x) * 8;
    float4 a = *(const float4*)(x + g);
    float4 b = *(const float4*)(x + g + 4);
    v8h h = { (_Float16)a.x, (_Float16)a.y, (_Float16)a.z, (_Float16)a.w,
              (_Float16)b.x, (_Float16)b.y, (_Float16)b.z, (_Float16)b.w };
    *(v8h*)(xh + g) = h;
}

// ----------------------------- expert_w f32 [e][k][n] -> f16 [e][n][k] (Wt)
// one block per 128x128 tile; grid = E * 8 * 8
__global__ void transpose_w_kernel(const float* __restrict__ W,
                                   _Float16* __restrict__ Wt) {
    __shared__ _Float16 tile[128][136];   // 272B row stride (16B aligned)
    int bid = blockIdx.x;
    int e   = bid >> 6;
    int r   = bid & 63;
    int kt  = r >> 3;
    int nt  = r & 7;
    const float* src = W + ((size_t)e << 20) + (size_t)(kt * 128) * HIDDEN + nt * 128;
    _Float16*    dst = Wt + ((size_t)e << 20) + (size_t)(nt * 128) * HIDDEN + kt * 128;

    int t = threadIdx.x;
    int col4 = (t & 31) * 4;
    int row0 = t >> 5;
#pragma unroll
    for (int p = 0; p < 16; ++p) {
        int row = row0 + p * 8;
        float4 f = *(const float4*)(src + (size_t)row * HIDDEN + col4);
        tile[col4 + 0][row] = (_Float16)f.x;
        tile[col4 + 1][row] = (_Float16)f.y;
        tile[col4 + 2][row] = (_Float16)f.z;
        tile[col4 + 3][row] = (_Float16)f.w;
    }
    __syncthreads();
    int n  = t >> 1;
    int kh = (t & 1) * 64;
    const v8h* srcl = (const v8h*)&tile[n][kh];
    v8h* dstg = (v8h*)(dst + (size_t)n * HIDDEN + kh);
#pragma unroll
    for (int j = 0; j < 8; ++j) dstg[j] = srcl[j];
}

// ------------------------------------------------- grouped expert GEMM (WMMA)
// grid.x = E * MT * NT  (MT = 64, NT = 8); block = 256 threads = 8 waves.
// Double-buffered LDS staged with GLOBAL_LOAD_ASYNC_TO_LDS_B128 (ASYNCcnt).
#define MT (NTOK / 128)
#define NT (HIDDEN / 128)
#define LDA 40                 // padded f16 row stride (80B)
#define STAGE_H (128 * LDA)    // 5120 halves = 10240 B per stage

__device__ __forceinline__ void async_copy16(unsigned lds_off, const void* gptr) {
    // per-lane 16B copy global -> LDS, tracked by ASYNCcnt
    asm volatile("global_load_async_to_lds_b128 %0, %1, off"
                 :: "v"(lds_off), "v"((unsigned long long)(uintptr_t)gptr)
                 : "memory");
}
__device__ __forceinline__ void wait_async0() {
    asm volatile("s_wait_asynccnt 0x0" ::: "memory");
}

__global__ void moe_gemm_kernel(const _Float16* __restrict__ tokens_h,
                                const _Float16* __restrict__ wt_h,
                                const int* __restrict__ counts,
                                const int* __restrict__ tok_list,
                                const float* __restrict__ wt_list,
                                float* __restrict__ out) {
    __shared__ _Float16 smem[4 * STAGE_H];   // A0 A1 B0 B1 = 40KB

    int bid = blockIdx.x;
    int e   = bid / (MT * NT);
    int rem = bid % (MT * NT);
    int mt  = rem / NT;
    int nt  = rem % NT;

    int cnt = counts[e];
    int m0  = mt * 128;
    if (m0 >= cnt) return;
    int n0  = nt * 128;

    const int*   tl = tok_list + (size_t)e * NTOK;
    const float* wl = wt_list + (size_t)e * NTOK;

    const int tid   = threadIdx.x;
    const int lane  = tid & 31;
    const int lhalf = lane >> 4;
    const int l16   = lane & 15;
    const int wave  = tid >> 5;
    const int m_off = (wave & 3) * 32;
    const int n_off = (wave >> 2) * 64;

    // staging role: thread -> (row 0..127, two 16B sub-chunks of the 64B row)
    const int s_row  = tid >> 1;
    const int s_sub0 = (tid & 1) * 2;     // subs {0,1} or {2,3}
    int a_tok = 0;
    {
        int gm = m0 + s_row;
        if (gm < cnt) a_tok = tl[gm];     // padded rows read token 0; discarded at store
    }
    const _Float16* a_base = tokens_h + (size_t)a_tok * HIDDEN;
    const _Float16* b_base = wt_h + ((size_t)e << 20) + (size_t)(n0 + s_row) * HIDDEN;
    const unsigned  a_lds0 = (unsigned)(s_row * (LDA * 2) + s_sub0 * 16);
    const unsigned  b_lds0 = a_lds0 + 2u * STAGE_H * 2u;   // B region starts at 20480B

    v8f c[2][4];
#pragma unroll
    for (int mi = 0; mi < 2; ++mi)
#pragma unroll
        for (int ni = 0; ni < 4; ++ni) c[mi][ni] = (v8f)0.f;

    // prologue: stage 0 (k0 = 0)
    {
        async_copy16(a_lds0,      a_base + s_sub0 * 8);
        async_copy16(a_lds0 + 16, a_base + s_sub0 * 8 + 8);
        async_copy16(b_lds0,      b_base + s_sub0 * 8);
        async_copy16(b_lds0 + 16, b_base + s_sub0 * 8 + 8);
    }

    for (int it = 0; it < HIDDEN / 32; ++it) {
        wait_async0();          // our stage-`it` copies have landed
        __syncthreads();        // everyone's copies landed; prev compute done

        if (it + 1 < HIDDEN / 32) {     // prefetch next stage into other buffer
            int k1 = (it + 1) * 32;
            unsigned so = (unsigned)(((it + 1) & 1) * (STAGE_H * 2));
            async_copy16(a_lds0 + so,      a_base + k1 + s_sub0 * 8);
            async_copy16(a_lds0 + so + 16, a_base + k1 + s_sub0 * 8 + 8);
            async_copy16(b_lds0 + so,      b_base + k1 + s_sub0 * 8);
            async_copy16(b_lds0 + so + 16, b_base + k1 + s_sub0 * 8 + 8);
        }

        const _Float16* As = smem + (it & 1) * STAGE_H;
        const _Float16* Bs = smem + 2 * STAGE_H + (it & 1) * STAGE_H;

        v16h a[2], b[4];
#pragma unroll
        for (int mi = 0; mi < 2; ++mi) {
            int ar = m_off + mi * 16 + l16;
            int ak = lhalf * 8;
            v8h lo = *(const v8h*)(As + ar * LDA + ak);
            v8h hi = *(const v8h*)(As + ar * LDA + ak + 16);
            a[mi] = __builtin_shufflevector(lo, hi, 0,1,2,3,4,5,6,7,8,9,10,11,12,13,14,15);
        }
#pragma unroll
        for (int ni = 0; ni < 4; ++ni) {
            int br = n_off + ni * 16 + l16;
            int bk = lhalf * 8;
            v8h lo = *(const v8h*)(Bs + br * LDA + bk);
            v8h hi = *(const v8h*)(Bs + br * LDA + bk + 16);
            b[ni] = __builtin_shufflevector(lo, hi, 0,1,2,3,4,5,6,7,8,9,10,11,12,13,14,15);
        }
#pragma unroll
        for (int mi = 0; mi < 2; ++mi)
#pragma unroll
            for (int ni = 0; ni < 4; ++ni)
                c[mi][ni] = __builtin_amdgcn_wmma_f32_16x16x32_f16(
                    false, a[mi], false, b[ni], (short)0, c[mi][ni], false, false);
    }

    // scatter-accumulate with gate weight applied per row (f32)
#pragma unroll
    for (int mi = 0; mi < 2; ++mi) {
#pragma unroll
        for (int r = 0; r < 8; ++r) {
            int rowin = mi * 16 + lhalf * 8 + r;     // C layout: M = r + 8*(lane/16)
            int gm2 = m0 + m_off + rowin;
            if (gm2 < cnt) {
                int   tok2 = tl[gm2];
                float wgt2 = wl[gm2];
                float* orow = out + (size_t)tok2 * HIDDEN + n0 + n_off + l16;
#pragma unroll
                for (int ni = 0; ni < 4; ++ni) {
                    __hip_atomic_fetch_add(orow + ni * 16, c[mi][ni][r] * wgt2,
                                           __ATOMIC_RELAXED, __HIP_MEMORY_SCOPE_AGENT);
                }
            }
        }
    }
}

// ------------------------------------------------------------------- launch
extern "C" void kernel_launch(void* const* d_in, const int* in_sizes, int n_in,
                              void* d_out, int out_size, void* d_ws, size_t ws_size,
                              hipStream_t stream) {
    const float* tokens   = (const float*)d_in[0];
    const float* router_w = (const float*)d_in[1];
    const float* router_b = (const float*)d_in[2];
    const float* expert_w = (const float*)d_in[3];
    float* out = (float*)d_out;

    // workspace layout
    float* logits  = (float*)d_ws;                         // T*E f32
    float* addend  = logits + (size_t)NTOK * NUM_EXPERTS;  // T f32
    int*   counts  = (int*)(addend + NTOK);                // 16 (+pad)
    int*   tok_lst = counts + 64;                          // E*T int
    float* wt_lst  = (float*)(tok_lst + (size_t)NUM_EXPERTS * NTOK);   // E*T f32
    _Float16* tok_h = (_Float16*)(wt_lst + (size_t)NUM_EXPERTS * NTOK); // T*H f16
    _Float16* w_h   = tok_h + (size_t)NTOK * HIDDEN;       // E*H*H f16 [e][n][k]

    zero_counts_kernel<<<1, 64, 0, stream>>>(counts);

    conv_tokens_kernel<<<(NTOK * HIDDEN) / (256 * 8), 256, 0, stream>>>(tokens, tok_h);

    transpose_w_kernel<<<NUM_EXPERTS * 64, 256, 0, stream>>>(expert_w, w_h);

    router_logits_kernel<<<(NTOK * NUM_EXPERTS) / 256, 256, 0, stream>>>(
        tokens, router_w, router_b, logits);

    route_kernel<<<NTOK / 256, 256, 0, stream>>>(
        logits, addend, counts, tok_lst, wt_lst);

    init_out_kernel<<<(NTOK * HIDDEN) / 256, 256, 0, stream>>>(out, addend);

    moe_gemm_kernel<<<NUM_EXPERTS * MT * NT, 256, 0, stream>>>(
        tok_h, w_h, counts, tok_lst, wt_lst, out);
}
